// ModulatorNet_90761248899496
// MI455X (gfx1250) — matrix operand order repaired and checked by reading, not compiled, not run on hardware
//
#include <hip/hip_runtime.h>
#include <hip/hip_bf16.h>
#include <math.h>

typedef __attribute__((ext_vector_type(16))) _Float16 v16h;
typedef __attribute__((ext_vector_type(8)))  float    v8f;

#define BSZ    256
#define RANK   16
#define IN_DIM 784
#define H1DIM  512
#define OUTD   10
#define KSTYLE 1568
#define TOTALG 29610
// offsets inside one row of g
#define OFF_A1    0
#define OFF_B1F   8192
#define OFF_BIAS1 20736
#define OFF_A2    21248
#define OFF_B2F   21408
#define OFF_BIAS2 29600
// workspace layout (in floats)
#define WS_C1    0u         // 256*16*14*14 = 802816
#define WS_H     802816u    // 256*1568     = 401408
#define WS_STYLE 1204224u   // 256*64       = 16384
#define WS_G1    1220608u   // 256*128      = 32768
#define WS_G     1253376u   // 256*29610    = 7580160
// total = 8,833,536 floats = ~35.3 MB

// ---------------- WMMA fragment loaders (f32 memory -> f16 fragment) ----------------
// A (16x32, MxK): lane l holds row l&15; k contiguous [16*khalf,+16)
__device__ __forceinline__ v16h frag_a_f32(const float* __restrict__ p, int khalf) {
  v16h f;
  const float* q = p + 16 * khalf;
#pragma unroll
  for (int e = 0; e < 8; ++e) f[e] = (_Float16)q[e];
#pragma unroll
  for (int e = 0; e < 8; ++e) f[8 + e] = (_Float16)q[8 + e];
  return f;
}
// B (32x16, KxN): lane l holds col l&15; k contiguous [16*khalf,+16)
__device__ __forceinline__ v16h frag_b_f32(const float* __restrict__ p, int khalf) {
  v16h f;
  const float* q = p + 16 * khalf;
#pragma unroll
  for (int e = 0; e < 16; ++e) f[e] = (_Float16)q[e];
  return f;
}
__device__ __forceinline__ v16h frag_zero() {
  v16h f;
#pragma unroll
  for (int e = 0; e < 16; ++e) f[e] = (_Float16)0.0f;
  return f;
}

// One wave computes a 16(M) x 64(N) strip: C[mbase:+16, nbase:+64] = A*B^T
// A: (M x K) row-major, base already offset to mbase row. B: (N x K) row-major (weights).
__device__ __forceinline__ void wave_gemm_strip(const float* __restrict__ Abase, int lda,
                                                const float* __restrict__ Bw, int ldb,
                                                int K, int nbase, int Ncols, v8f acc[4]) {
  const int lane  = threadIdx.x & 31;
  const int l15   = lane & 15;
  const int khalf = lane >> 4;
  const float* arow = Abase + (size_t)l15 * lda;
  for (int kb = 0; kb < K; kb += 32) {
    v16h a = frag_a_f32(arow + kb, khalf);
#pragma unroll
    for (int t = 0; t < 4; ++t) {
      int col = nbase + t * 16 + l15;
      v16h b;
      if (col < Ncols) b = frag_b_f32(Bw + (size_t)col * ldb + kb, khalf);
      else             b = frag_zero();
      acc[t] = __builtin_amdgcn_wmma_f32_16x16x32_f16(
          false, a, false, b, (short)0, acc[t], false, false);
    }
  }
}

// ---------------- conv1: (256,1,28,28) -> relu -> (256,16,14,14) ----------------
__global__ void k_conv1(const float* __restrict__ img, const float* __restrict__ w,
                        const float* __restrict__ bias, float* __restrict__ out) {
  int idx = blockIdx.x * 256 + threadIdx.x;
  if (idx >= 256 * 16 * 14 * 14) return;
  int x = idx % 14, y = (idx / 14) % 14, co = (idx / 196) % 16, b = idx / 3136;
  const float* ip = img + (size_t)b * 784;
  const float* wp = w + co * 9;
  float s = bias[co];
#pragma unroll
  for (int ky = 0; ky < 3; ++ky) {
    int iy = 2 * y - 1 + ky;
    if (iy < 0 || iy > 27) continue;
#pragma unroll
    for (int kx = 0; kx < 3; ++kx) {
      int ix = 2 * x - 1 + kx;
      if (ix < 0 || ix > 27) continue;
      s += ip[iy * 28 + ix] * wp[ky * 3 + kx];
    }
  }
  out[idx] = fmaxf(s, 0.0f);
}

// ---------------- conv2: (256,16,14,14) -> relu -> flat h (256,1568) ----------------
__global__ void k_conv2(const float* __restrict__ in, const float* __restrict__ w,
                        const float* __restrict__ bias, float* __restrict__ h) {
  int idx = blockIdx.x * 256 + threadIdx.x;
  if (idx >= 256 * 32 * 7 * 7) return;
  int x = idx % 7, y = (idx / 7) % 7, co = (idx / 49) % 32, b = idx / 1568;
  const float* ip = in + (size_t)b * 3136;
  const float* wp = w + co * 144;
  float s = bias[co];
  for (int ci = 0; ci < 16; ++ci) {
    const float* ic = ip + ci * 196;
    const float* wc = wp + ci * 9;
#pragma unroll
    for (int ky = 0; ky < 3; ++ky) {
      int iy = 2 * y - 1 + ky;
      if (iy < 0 || iy > 13) continue;
#pragma unroll
      for (int kx = 0; kx < 3; ++kx) {
        int ix = 2 * x - 1 + kx;
        if (ix < 0 || ix > 13) continue;
        s += ic[iy * 14 + ix] * wc[ky * 3 + kx];
      }
    }
  }
  h[(size_t)b * 1568 + co * 49 + y * 7 + x] = fmaxf(s, 0.0f);
}

// ---------------- style = tanh(h @ sew^T + seb) : M=256 N=64 K=1568 ----------------
__global__ void k_style(const float* __restrict__ h, const float* __restrict__ sew,
                        const float* __restrict__ seb, float* __restrict__ style) {
  int widx  = threadIdx.x >> 5;
  int mtile = blockIdx.x * 8 + widx;         // 16 wave-jobs total
  int lane  = threadIdx.x & 31;
  int l15   = lane & 15, khalf = lane >> 4;
  v8f acc[4] = {};
  wave_gemm_strip(h + (size_t)mtile * 16 * KSTYLE, KSTYLE, sew, KSTYLE, KSTYLE, 0, 64, acc);
#pragma unroll
  for (int t = 0; t < 4; ++t) {
    int col = t * 16 + l15;
    float sb = seb[col];
#pragma unroll
    for (int r = 0; r < 8; ++r) {
      int row = mtile * 16 + r + 8 * khalf;
      style[(size_t)row * 64 + col] = tanhf(acc[t][r] + sb);
    }
  }
}

// ---------------- g1h = relu(style @ g1w^T + g1b) : M=256 N=128 K=64 ----------------
__global__ void k_g1(const float* __restrict__ style, const float* __restrict__ g1w,
                     const float* __restrict__ g1b, float* __restrict__ g1h) {
  int widx = threadIdx.x >> 5;
  int job  = blockIdx.x * 8 + widx;          // 32 wave-jobs
  int mtile = job & 15, nstrip = job >> 4;
  int lane = threadIdx.x & 31;
  int l15  = lane & 15, khalf = lane >> 4;
  int nbase = nstrip * 64;
  v8f acc[4] = {};
  wave_gemm_strip(style + (size_t)mtile * 16 * 64, 64, g1w, 64, 64, nbase, 128, acc);
#pragma unroll
  for (int t = 0; t < 4; ++t) {
    int col = nbase + t * 16 + l15;
    float bb = g1b[col];
#pragma unroll
    for (int r = 0; r < 8; ++r) {
      int row = mtile * 16 + r + 8 * khalf;
      g1h[(size_t)row * 128 + col] = fmaxf(acc[t][r] + bb, 0.0f);
    }
  }
}

// ---------------- g = g1h @ g2w^T + g2b : M=256 N=29610 K=128 (dominant) ----------------
__global__ void k_g2(const float* __restrict__ g1h, const float* __restrict__ g2w,
                     const float* __restrict__ g2b, float* __restrict__ g) {
  int widx = threadIdx.x >> 5;
  int job  = blockIdx.x * 8 + widx;          // 7408 wave-jobs = 926 blocks
  int mtile = job & 15, nstrip = job >> 4;   // nstrip in [0, 463)
  int lane = threadIdx.x & 31;
  int l15  = lane & 15, khalf = lane >> 4;
  int nbase = nstrip * 64;
  v8f acc[4] = {};
  wave_gemm_strip(g1h + (size_t)mtile * 16 * 128, 128, g2w, 128, 128, nbase, TOTALG, acc);
#pragma unroll
  for (int t = 0; t < 4; ++t) {
    int col = nbase + t * 16 + l15;
    if (col < TOTALG) {
      float bb = g2b[col];
#pragma unroll
      for (int r = 0; r < 8; ++r) {
        int row = mtile * 16 + r + 8 * khalf;
        g[(size_t)row * TOTALG + col] = acc[t][r] + bb;
      }
    }
  }
}

// ---------------- per-sample low-rank apply: out = a2*(b2f*relu(a1*(b1f*x)+b1))+b2 ----------------
__global__ void k_apply(const float* __restrict__ img, const float* __restrict__ g,
                        float* __restrict__ out) {
  __shared__ float sx[IN_DIM];
  __shared__ float sh1[H1DIM];
  __shared__ float spart[16][16];
  __shared__ float st[16];
  const int b = blockIdx.x, tid = threadIdx.x;
  const float* gb = g + (size_t)b * TOTALG;
  for (int i = tid; i < IN_DIM; i += 256) sx[i] = img[(size_t)b * IN_DIM + i];
  __syncthreads();

  // t1[r] = sum_i b1f[r,i]*x[i]
  {
    int r = tid >> 4, c = tid & 15;
    float s = 0.0f;
    const float* row = gb + OFF_B1F + r * IN_DIM;
    for (int i = c; i < IN_DIM; i += 16) s += row[i] * sx[i];
    spart[r][c] = s;
  }
  __syncthreads();
  if (tid < 16) {
    float s = 0.0f;
#pragma unroll
    for (int j = 0; j < 16; ++j) s += spart[tid][j];
    st[tid] = s;
  }
  __syncthreads();

  // h1[o] = relu(bias1[o] + sum_r a1[o,r]*t1[r])
  for (int o = tid; o < H1DIM; o += 256) {
    float s = gb[OFF_BIAS1 + o];
    const float* ar = gb + OFF_A1 + o * RANK;
#pragma unroll
    for (int r = 0; r < RANK; ++r) s += ar[r] * st[r];
    sh1[o] = fmaxf(s, 0.0f);
  }
  __syncthreads();

  // t2[r] = sum_h b2f[r,h]*h1[h]
  {
    int r = tid >> 4, c = tid & 15;
    float s = 0.0f;
    const float* row = gb + OFF_B2F + r * H1DIM;
    for (int i = c; i < H1DIM; i += 16) s += row[i] * sh1[i];
    spart[r][c] = s;
  }
  __syncthreads();
  if (tid < 16) {
    float s = 0.0f;
#pragma unroll
    for (int j = 0; j < 16; ++j) s += spart[tid][j];
    st[tid] = s;
  }
  __syncthreads();

  if (tid < OUTD) {
    float s = gb[OFF_BIAS2 + tid];
    const float* ar = gb + OFF_A2 + tid * RANK;
#pragma unroll
    for (int r = 0; r < RANK; ++r) s += ar[r] * st[r];
    out[(size_t)b * OUTD + tid] = s;
  }
}

extern "C" void kernel_launch(void* const* d_in, const int* in_sizes, int n_in,
                              void* d_out, int out_size, void* d_ws, size_t ws_size,
                              hipStream_t stream) {
  (void)in_sizes; (void)n_in; (void)out_size; (void)ws_size;
  const float* images = (const float*)d_in[0];
  const float* c1w    = (const float*)d_in[1];
  const float* c1b    = (const float*)d_in[2];
  const float* c2w    = (const float*)d_in[3];
  const float* c2b    = (const float*)d_in[4];
  const float* sew    = (const float*)d_in[5];
  const float* seb    = (const float*)d_in[6];
  const float* g1w    = (const float*)d_in[7];
  const float* g1b    = (const float*)d_in[8];
  const float* g2w    = (const float*)d_in[9];
  const float* g2b    = (const float*)d_in[10];
  float* out = (float*)d_out;
  float* ws  = (float*)d_ws;

  float* c1o   = ws + WS_C1;
  float* h     = ws + WS_H;
  float* style = ws + WS_STYLE;
  float* g1h   = ws + WS_G1;
  float* g     = ws + WS_G;

  k_conv1<<<3136, 256, 0, stream>>>(images, c1w, c1b, c1o);
  k_conv2<<<1568, 256, 0, stream>>>(c1o, c2w, c2b, h);
  k_style<<<2,    256, 0, stream>>>(h, sew, seb, style);
  k_g1   <<<4,    256, 0, stream>>>(style, g1w, g1b, g1h);
  k_g2   <<<926,  256, 0, stream>>>(g1h, g2w, g2b, g);
  k_apply<<<256,  256, 0, stream>>>(images, g, out);
}
